// NetworkBasic_52982716563855
// MI455X (gfx1250) — matrix operand order repaired and checked by reading, not compiled, not run on hardware
//
#include <hip/hip_runtime.h>
#include <math.h>

// SLAYER SNN on MI455X (gfx1250, wave32).
//   psp(tau) is a linear recurrence == strictly-lower-triangular 64x64 temporal
//   matmul: out[t] = sum_{s<t} c*(t-s)*d^(t-s) * x[s].  -> v_wmma_f32_16x16x32_f16
//   LIF is nonlinear (spikes feed the refractory state) -> sequential t-loop,
//   fused with the spatial conv via LDS tiles, state (u,v) in registers.
// Traffic is L2-resident (192MB L2 >> 113MB of intermediates); only the
// compulsory ~75MB touches HBM (~3.2us at 23.3 TB/s).

typedef __attribute__((ext_vector_type(16))) _Float16 v16h;
typedef __attribute__((ext_vector_type(8)))  float    v8f;

#define NIMG 4
#define IMH  192
#define IMW  192
#define NT   64
#define NPIX (NIMG * IMH * IMW)

// ---------------------------------------------------------------------------
// psp stage: OUT(64 x 16px) = L(64x64, strictly lower triangular) @ X(64 x 16px)
// per wave, using 4 M-tiles x 2 K-chunks of v_wmma_f32_16x16x32_f16.
// x   : (P, T) float32, time-contiguous (matches (N,C,H,W,T) flat layout)
// out : (T, P) float32 (transposed so the conv stage reads spatially contiguous)
// ---------------------------------------------------------------------------
__global__ __launch_bounds__(256) void psp_wmma_kernel(
    const float* __restrict__ x,
    float* __restrict__ out,
    float c, float inv_tau)
{
  __shared__ float wtab[NT];               // wtab[k] = c * k * exp(-k/tau), wtab[0]=0
  if (threadIdx.x < NT) {
    const int k = threadIdx.x;
    wtab[k] = (k > 0) ? c * (float)k * __expf(-(float)k * inv_tau) : 0.0f;
  }
  __syncthreads();

  const int  lane   = threadIdx.x & 31;
  const int  laneLo = lane & 15;
  const bool hiHalf = lane >= 16;
  const int  wave   = blockIdx.x * (blockDim.x >> 5) + (threadIdx.x >> 5);
  const int  p      = wave * 16 + laneLo;  // this lane's pixel (column of B/D)

  // B fragments (X): 32x16 f16, lane%16 = pixel column; lanes<16 hold K 0..15,
  // lanes>=16 hold K 16..31 of the K-chunk, 2 halfs per VGPR (element e == K).
  v16h bfrag[2];
#pragma unroll
  for (int kc = 0; kc < 2; ++kc) {
    const float* src = x + (size_t)p * NT + kc * 32 + (hiHalf ? 16 : 0);
#pragma unroll
    for (int e = 0; e < 16; ++e) bfrag[kc][e] = (_Float16)src[e];
  }

  v8f acc[4] = {};                         // 4 M-tiles of 16 output timesteps

#pragma unroll
  for (int kc = 0; kc < 2; ++kc) {
#pragma unroll
    for (int mt = 0; mt < 4; ++mt) {
      // A fragment (L tile): 16x32 f16. lane<16: M=lane, K in {0..15};
      // lane>=16: M=lane-16, K in {16..31}; element e -> K = 16*(lane>=16) + e.
      v16h a;
      const int gm = mt * 16 + laneLo;
#pragma unroll
      for (int e = 0; e < 16; ++e) {
        const int kl = (hiHalf ? 16 : 0) + e;
        int dlt = gm - (kc * 32 + kl);
        if (dlt < 0) dlt = 0;              // wtab[0] == 0 handles s >= t
        a[e] = (_Float16)wtab[dlt];
      }
      acc[mt] = __builtin_amdgcn_wmma_f32_16x16x32_f16(
          false, a, false, bfrag[kc], (short)0, acc[mt], false, false);
    }
  }

  // D layout: VGPR i -> (M=i, N=lane) for lanes<16, (M=8+i, N=lane-16) otherwise.
#pragma unroll
  for (int mt = 0; mt < 4; ++mt)
#pragma unroll
    for (int i = 0; i < 8; ++i) {
      const int t = mt * 16 + (hiHalf ? 8 + i : i);
      out[(size_t)t * NPIX + p] = acc[mt][i];
    }
}

// ---------------------------------------------------------------------------
// Fused per-timestep spatial conv (KSxKS, zero pad KS/2) + LIF with alpha
// refractory recurrence. One thread per output pixel of a 16x16 tile; the
// t-loop is the inherently serial part (spikes feed the refractory state).
// psp  : (T, P) float32
// sout : (P, T) float32 spikes (amplitude 1/Ts, Ts = 1)
// ---------------------------------------------------------------------------
template <int KS>
__global__ __launch_bounds__(256) void conv_lif_kernel(
    const float* __restrict__ psp,
    const float* __restrict__ wk,        // (KS*KS) conv weights (cross-correlation)
    float* __restrict__ sout,
    float d, float cref, float theta)
{
  constexpr int TB   = 16;
  constexpr int HALO = KS / 2;
  constexpr int S    = TB + 2 * HALO;
  __shared__ float tile[S][S + 1];
  __shared__ float wsh[KS * KS];

  const int tx = threadIdx.x & (TB - 1);
  const int ty = threadIdx.x >> 4;
  constexpr int NBX = IMW / TB;
  constexpr int NBY = IMH / TB;
  const int bx = blockIdx.x % NBX;
  const int by = (blockIdx.x / NBX) % NBY;
  const int n  = blockIdx.x / (NBX * NBY);
  const int h0 = by * TB, w0 = bx * TB;
  const int gh = h0 + ty, gw = w0 + tx;
  const size_t imgbase = (size_t)n * IMH * IMW;
  const size_t pix     = imgbase + (size_t)gh * IMW + gw;

  if (threadIdx.x < KS * KS) wsh[threadIdx.x] = wk[threadIdx.x];

  float u = 0.0f, v = 0.0f;                // refractory recurrence state
  for (int t = 0; t < NT; ++t) {
    const float* src = psp + (size_t)t * NPIX + imgbase;
    __syncthreads();                       // guards prior-iter reads (+wsh at t==0)
    for (int i = threadIdx.x; i < S * S; i += 256) {
      const int ly = i / S, lx = i % S;
      const int hh = h0 + ly - HALO, ww = w0 + lx - HALO;
      tile[ly][lx] = (hh >= 0 && hh < IMH && ww >= 0 && ww < IMW)
                         ? src[(size_t)hh * IMW + ww] : 0.0f;
    }
    __syncthreads();

    float accv = 0.0f;
#pragma unroll
    for (int ky = 0; ky < KS; ++ky)
#pragma unroll
      for (int kx = 0; kx < KS; ++kx)
        accv = fmaf(wsh[ky * KS + kx], tile[ty + ky][tx + kx], accv);

    // LIF step: v_new from past spikes only; m = conv + c*v_new; threshold.
    const float v_new = d * (v + u);
    const float m = accv + cref * v_new;
    const float s = (m >= theta) ? 1.0f : 0.0f;
    u = d * u + s;
    v = v_new;
    sout[pix * NT + t] = s;                // s / Ts, Ts = 1
  }
}

// ---------------------------------------------------------------------------
// d_in[0] = spikeInput (4,1,192,192,64) f32   d_in[1] = W1 (5x5)   d_in[2] = W2 (3x3)
// d_out   = s2 (4,1,192,192,64) f32
// d_ws    = one (T,P) f32 buffer (37.75 MB), reused for psp1 then psp2.
// d_out doubles as the s1 scratch between stages 2 and 3 (fully rewritten by 4).
// ---------------------------------------------------------------------------
extern "C" void kernel_launch(void* const* d_in, const int* in_sizes, int n_in,
                              void* d_out, int out_size, void* d_ws, size_t ws_size,
                              hipStream_t stream) {
  (void)in_sizes; (void)n_in; (void)out_size; (void)ws_size;
  const float* x  = (const float*)d_in[0];
  const float* W1 = (const float*)d_in[1];
  const float* W2 = (const float*)d_in[2];
  float* out = (float*)d_out;
  float* psp = (float*)d_ws;

  const float E  = 2.71828182845904523f;
  const float D1 = 0.36787944117144233f;   // exp(-1)   (tauSr/tauRef = 1)
  const float D2 = 0.60653065971263342f;   // exp(-1/2) (tauSr/tauRef = 2)

  dim3 blk(256);
  dim3 gpsp(NPIX / 128);                               // 8 waves x 16 px each
  dim3 gconv((IMW / 16) * (IMH / 16) * NIMG);

  // stage 1: psp1 = L(tau=1) @ x            c = e*Ts^2/tau = e
  psp_wmma_kernel<<<gpsp, blk, 0, stream>>>(x, psp, E, 1.0f);
  // stage 2: conv5x5 + LIF(theta=30, tauRef=1): cref = -scaleRef*theta*e/tau = -30e
  conv_lif_kernel<5><<<gconv, blk, 0, stream>>>(psp, W1, out, D1, -30.0f * E, 30.0f);
  // stage 3: psp2 = L(tau=2) @ s1           c = e/2
  psp_wmma_kernel<<<gpsp, blk, 0, stream>>>(out, psp, 0.5f * E, 0.5f);
  // stage 4: conv3x3 + LIF(theta=50, tauRef=2): cref = -50e/2 = -25e
  conv_lif_kernel<3><<<gconv, blk, 0, stream>>>(psp, W2, out, D2, -25.0f * E, 50.0f);
}
